// PSFNet_46213848105463
// MI455X (gfx1250) — matrix operand order repaired and checked by reading, not compiled, not run on hardware
//
#include <hip/hip_runtime.h>
#include <cstddef>

// ---------------------------------------------------------------------------
// PSFNet on MI455X (gfx1250): bf16 WMMA for the 13 shared-input MLP GEMMs,
// VALU float4 chord-mix (L2-resident V), deterministic two-pass final GEMV.
// Gelu is branch-free (A&S 7.1.26 erf, |err|<1.5e-7 << bf16 rounding of H).
// ---------------------------------------------------------------------------

#define DEV_INLINE __device__ __forceinline__

typedef __attribute__((ext_vector_type(16))) __bf16 v16bf;
typedef __attribute__((ext_vector_type(8)))  __bf16 v8bf;
typedef __attribute__((ext_vector_type(2)))  __bf16 v2bf;
typedef __attribute__((ext_vector_type(8)))  float  v8f;

constexpr int B_   = 32;
constexpr int N_   = 4096;
constexpr int E_   = 128;
constexpr int NW_  = 12;
constexpr int NL_  = 13;
constexpr int C_   = 128;
constexpr int NC_  = 10;
constexpr int ROWS = B_ * N_;      // 131072
constexpr int NSET = NW_ + 1;      // 13 weight sets: [g, fs0..fs11]

// ---- workspace layout (bytes) ---------------------------------------------
constexpr size_t OFF_X    = 0;                      // bf16 x   (ROWS,128)  32 MB
constexpr size_t OFF_W1P  = OFF_X    + (size_t)ROWS * E_ * 2;          // 13*128*128 bf16 packed
constexpr size_t OFF_W2P  = OFF_W1P  + (size_t)NSET * 128 * 128 * 2;
constexpr size_t OFF_B1P  = OFF_W2P  + (size_t)NSET * 128 * 128 * 2;   // 13*128 f32
constexpr size_t OFF_B2P  = OFF_B1P  + (size_t)NSET * 128 * 4;
constexpr size_t OFF_WM   = OFF_B2P  + (size_t)NSET * 128 * 4;         // (ROWS,16) f32
constexpr size_t OFF_R    = OFF_WM   + (size_t)ROWS * 16 * 4;          // res = V0   64 MB
constexpr size_t OFF_VA   = OFF_R    + (size_t)ROWS * C_ * 4;          // ping       64 MB
constexpr size_t OFF_VB   = OFF_VA   + (size_t)ROWS * C_ * 4;          // pong       64 MB
constexpr size_t OFF_PART = OFF_VB   + (size_t)ROWS * C_ * 4;          // 32*128*10 f32

// ---- WMMA fragment helpers (layouts per CDNA5 ISA §7.12.2) -----------------
union AFrag { v16bf v; struct { v8bf lo, hi; } h; };

// 16x32 bf16 A tile from a row-major buffer (stride ld elements).
// lane<16 : M=lane,    K = {0..7, 16..23}
// lane>=16: M=lane-16, K = {8..15, 24..31}
DEV_INLINE v16bf load_a_frag(const __bf16* tile, int ld, int lane) {
  const int m  = lane & 15;
  const int kb = (lane & 16) ? 8 : 0;
  const __bf16* p = tile + m * ld + kb;
  AFrag f;
  f.h.lo = *(const v8bf*)(p);
  f.h.hi = *(const v8bf*)(p + 16);
  return f.v;
}

DEV_INLINE v8f wmma_bf16(v16bf a, v16bf b, v8f c) {
  return __builtin_amdgcn_wmma_f32_16x16x32_bf16(
      /*neg_a=*/false, a, /*neg_b=*/false, b,
      /*c_mod=*/(short)0, c, /*reuse_a=*/false, /*reuse_b=*/false);
}

// e^a, branch-free: v_exp_f32 computes 2^x.
DEV_INLINE float fast_exp(float a) {
  return __builtin_amdgcn_exp2f(a * 1.4426950408889634f);
}

// Exact-form gelu 0.5x(1+erf(x/sqrt(2))) with branch-free erf
// (Abramowitz & Stegun 7.1.26, |err| < 1.5e-7): 5 FMAs + v_rcp + v_exp.
DEV_INLINE float gelu_exact(float x) {
  const float z  = x * 0.70710678118654752f;
  const float az = fabsf(z);
  const float t  = __builtin_amdgcn_rcpf(fmaf(0.3275911f, az, 1.0f));
  float p = 1.061405429f;
  p = fmaf(p, t, -1.453152027f);
  p = fmaf(p, t,  1.421413741f);
  p = fmaf(p, t, -0.284496736f);
  p = fmaf(p, t,  0.254829592f);
  p = p * t;
  const float e = fast_exp(-az * az);
  float erfv = fmaf(-p, e, 1.0f);       // erf(|z|)
  erfv = copysignf(erfv, z);
  return 0.5f * x * (1.0f + erfv);
}

// ---------------------------------------------------------------------------
// Pack W1/W2 (fp32 -> bf16) into B-fragment layout:
//   per matrix s: tile (kt,nt) -> 32 lanes x 16 elems, elem e: K=kt*32+kBase+(e&7)+(e&8?16:0),
//   N=nt*16+(lane&15), kBase = lane<16 ? 0 : 8.  Also pack biases (fs_b2 pad->16).
// ---------------------------------------------------------------------------
__global__ void pack_kernel(const float* __restrict__ gW1, const float* __restrict__ fsW1,
                            const float* __restrict__ gW2, const float* __restrict__ fsW2,
                            const float* __restrict__ gb1, const float* __restrict__ fsb1,
                            const float* __restrict__ gb2, const float* __restrict__ fsb2,
                            __bf16* __restrict__ w1p, __bf16* __restrict__ w2p,
                            float* __restrict__ b1p, float* __restrict__ b2p) {
  const int idx = blockIdx.x * blockDim.x + threadIdx.x;
  const int TOT = NSET * 4 * 8 * 32 * 16;   // 212992 packed elements per array
  if (idx < TOT) {
    const int e = idx & 15;
    const int l = (idx >> 4) & 31;
    const int u = (idx >> 9) & 7;           // n-tile
    const int t = (idx >> 12) & 3;          // k-tile
    const int s = idx >> 14;                // weight set
    const int kb = (l & 16) ? 8 : 0;
    const int k = t * 32 + kb + (e & 7) + ((e & 8) << 1);
    const int n = u * 16 + (l & 15);
    const float w1 = (s == 0) ? gW1[k * 128 + n]
                              : fsW1[((size_t)(s - 1) * 128 + k) * 128 + n];
    float w2;
    if (s == 0) w2 = gW2[k * 128 + n];
    else        w2 = (n < NL_) ? fsW2[((size_t)(s - 1) * 128 + k) * NL_ + n] : 0.0f;
    w1p[idx] = (__bf16)w1;
    w2p[idx] = (__bf16)w2;
  }
  if (idx < NSET * 128) {
    const int s = idx >> 7, j = idx & 127;
    b1p[idx] = (s == 0) ? gb1[j] : fsb1[(s - 1) * 128 + j];
    float b2;
    if (s == 0) b2 = gb2[j];
    else        b2 = (j < NL_) ? fsb2[(s - 1) * NL_ + j] : 0.0f;
    b2p[idx] = b2;
  }
}

// ---------------------------------------------------------------------------
// x[row, :] = bf16( emb[tok[row], :] + pos_emb[row % N, :] )
// ---------------------------------------------------------------------------
__global__ void embed_kernel(const int* __restrict__ tokens, const float* __restrict__ emb,
                             const float* __restrict__ pos, __bf16* __restrict__ x) {
  const int tid = blockIdx.x * blockDim.x + threadIdx.x;  // ROWS*64
  const int row = tid >> 6;
  const int j   = (tid & 63) * 2;
  const int tok = tokens[row];
  const int n   = row & (N_ - 1);
  const float a0 = emb[tok * E_ + j]     + pos[n * E_ + j];
  const float a1 = emb[tok * E_ + j + 1] + pos[n * E_ + j + 1];
  v2bf o; o.x = (__bf16)a0; o.y = (__bf16)a1;
  *(v2bf*)(x + (size_t)row * E_ + j) = o;
}

// ---------------------------------------------------------------------------
// Fused MLP: out = gelu(x @ W1 + b1) @ W2 + b2 for weight set s.
// One wave per 16-row tile; H round-trips through a private LDS slice (same-wave
// DS ordering). nOutTiles=8/outLd=128 for the gate, 1/16 for fs (cols>=13 are 0).
// ---------------------------------------------------------------------------
__global__ __launch_bounds__(256) void mlp_kernel(
    const __bf16* __restrict__ x, const __bf16* __restrict__ w1pAll,
    const __bf16* __restrict__ w2pAll, const float* __restrict__ b1pAll,
    const float* __restrict__ b2pAll, float* __restrict__ out,
    int s, int nOutTiles, int outLd) {
  __shared__ __bf16 sH[8][16 * 128];                 // 32 KB: 4 KB per wave
  const int lane = threadIdx.x & 31;
  const int wave = threadIdx.x >> 5;
  const int row0 = (blockIdx.x * 8 + wave) * 16;

  const __bf16* w1p = w1pAll + (size_t)s * 16384;
  const __bf16* w2p = w2pAll + (size_t)s * 16384;
  const float*  b1  = b1pAll + s * 128;
  const float*  b2  = b2pAll + s * 128;

  const int colL = lane & 15;
  const int rOff = (lane & 16) ? 8 : 0;

  // ---- phase 1: H = gelu(x @ W1 + b1), 16 x 128 ----
  v16bf a[4];
  #pragma unroll
  for (int t = 0; t < 4; ++t)
    a[t] = load_a_frag(x + (size_t)row0 * E_ + t * 32, E_, lane);

  __bf16* myH = &sH[wave][0];
  #pragma unroll
  for (int nt = 0; nt < 8; ++nt) {
    const float bv = b1[nt * 16 + colL];
    v8f acc;
    #pragma unroll
    for (int r = 0; r < 8; ++r) acc[r] = bv;
    #pragma unroll
    for (int t = 0; t < 4; ++t) {
      v16bf bfrag = *(const v16bf*)(w1p + ((t * 8 + nt) * 32 + lane) * 16);
      acc = wmma_bf16(a[t], bfrag, acc);
    }
    #pragma unroll
    for (int r = 0; r < 8; ++r)
      myH[(r + rOff) * 128 + nt * 16 + colL] = (__bf16)gelu_exact(acc[r]);
  }

  // ---- phase 2: out = H @ W2 + b2 ----
  v16bf ah[4];
  #pragma unroll
  for (int t = 0; t < 4; ++t)
    ah[t] = load_a_frag(myH + t * 32, 128, lane);

  for (int nt = 0; nt < nOutTiles; ++nt) {
    const float bv = b2[nt * 16 + colL];
    v8f acc;
    #pragma unroll
    for (int r = 0; r < 8; ++r) acc[r] = bv;
    #pragma unroll
    for (int t = 0; t < 4; ++t) {
      v16bf bfrag = *(const v16bf*)(w2p + ((t * 8 + nt) * 32 + lane) * 16);
      acc = wmma_bf16(ah[t], bfrag, acc);
    }
    #pragma unroll
    for (int r = 0; r < 8; ++r)
      out[(size_t)(row0 + r + rOff) * outLd + nt * 16 + colL] = acc[r];
  }
}

// ---------------------------------------------------------------------------
// Chord mix: Vout[b,n,:] = res[b,n,:] + sum_k Wm[b,n,k] * Vin[b, col(n,k), :]
// col(n,0)=n, col(n,k)=(n+2^(k-1)) mod N.  One wave per row (float4/lane).
// V (64MB) is L2-resident, so the 13-way gather stays on-die.
// ---------------------------------------------------------------------------
__global__ __launch_bounds__(256) void chord_kernel(
    const float4* __restrict__ Vin, const float4* __restrict__ res,
    const float4* __restrict__ Wm, float4* __restrict__ Vout) {
  const int lane = threadIdx.x & 31;
  const int row  = blockIdx.x * 8 + (threadIdx.x >> 5);
  const int b    = row >> 12;
  const int n    = row & (N_ - 1);
  float4 acc = res[(size_t)row * 32 + lane];
  // 13 mixing weights as 4 x b128 loads (last 3 lanes of w[3] are zero/unused)
  const float4 w4[4] = { Wm[(size_t)row * 4 + 0], Wm[(size_t)row * 4 + 1],
                         Wm[(size_t)row * 4 + 2], Wm[(size_t)row * 4 + 3] };
  const float* w = (const float*)&w4[0];
  #pragma unroll
  for (int k = 0; k < NL_; ++k) {
    const int col = (k == 0) ? n : ((n + (1 << (k - 1))) & (N_ - 1));
    const float wk = w[k];
    const float4 v = Vin[((size_t)b * N_ + col) * 32 + lane];
    acc.x += wk * v.x; acc.y += wk * v.y; acc.z += wk * v.z; acc.w += wk * v.w;
  }
  Vout[(size_t)row * 32 + lane] = acc;
}

// ---------------------------------------------------------------------------
// Final: out(32,10) = V.reshape(32, N*C) @ final_W + final_b  (split-K, 2 pass)
// ---------------------------------------------------------------------------
__global__ __launch_bounds__(256) void final_partial_kernel(
    const float* __restrict__ V, const float* __restrict__ W,
    float* __restrict__ partial) {
  __shared__ float red[NC_][256];
  const int b    = blockIdx.y;
  const int base = blockIdx.x * 4096;
  const int tid  = threadIdx.x;
  float acc[NC_];
  #pragma unroll
  for (int o = 0; o < NC_; ++o) acc[o] = 0.0f;
  for (int it = 0; it < 16; ++it) {
    const int i = base + it * 256 + tid;
    const float v = V[(size_t)b * (N_ * C_) + i];
    const float* wr = W + (size_t)i * NC_;
    #pragma unroll
    for (int o = 0; o < NC_; ++o) acc[o] += v * wr[o];
  }
  #pragma unroll
  for (int o = 0; o < NC_; ++o) red[o][tid] = acc[o];
  __syncthreads();
  for (int st = 128; st > 0; st >>= 1) {
    if (tid < st) {
      #pragma unroll
      for (int o = 0; o < NC_; ++o) red[o][tid] += red[o][tid + st];
    }
    __syncthreads();
  }
  if (tid < NC_)
    partial[((size_t)b * gridDim.x + blockIdx.x) * NC_ + tid] = red[tid][0];
}

__global__ void final_reduce_kernel(const float* __restrict__ partial,
                                    const float* __restrict__ bias,
                                    float* __restrict__ out, int chunks) {
  const int tid = threadIdx.x;
  if (tid >= B_ * NC_) return;
  const int b = tid / NC_, o = tid % NC_;
  float s = bias[o];
  for (int c = 0; c < chunks; ++c)
    s += partial[((size_t)b * chunks + c) * NC_ + o];
  out[tid] = s;
}

// ---------------------------------------------------------------------------
extern "C" void kernel_launch(void* const* d_in, const int* in_sizes, int n_in,
                              void* d_out, int out_size, void* d_ws, size_t ws_size,
                              hipStream_t stream) {
  (void)in_sizes; (void)n_in; (void)out_size; (void)ws_size;
  const int*   tokens  = (const int*)  d_in[0];
  const float* emb     = (const float*)d_in[1];
  const float* pos_emb = (const float*)d_in[2];
  const float* fs_W1   = (const float*)d_in[3];
  const float* fs_b1   = (const float*)d_in[4];
  const float* fs_W2   = (const float*)d_in[5];
  const float* fs_b2   = (const float*)d_in[6];
  const float* g_W1    = (const float*)d_in[7];
  const float* g_b1    = (const float*)d_in[8];
  const float* g_W2    = (const float*)d_in[9];
  const float* g_b2    = (const float*)d_in[10];
  const float* final_W = (const float*)d_in[11];
  const float* final_b = (const float*)d_in[12];
  float* out = (float*)d_out;

  char* ws = (char*)d_ws;
  __bf16* x    = (__bf16*)(ws + OFF_X);
  __bf16* w1p  = (__bf16*)(ws + OFF_W1P);
  __bf16* w2p  = (__bf16*)(ws + OFF_W2P);
  float*  b1p  = (float*) (ws + OFF_B1P);
  float*  b2p  = (float*) (ws + OFF_B2P);
  float*  Wm   = (float*) (ws + OFF_WM);
  float*  R    = (float*) (ws + OFF_R);    // V0 == residual
  float*  VA   = (float*) (ws + OFF_VA);
  float*  VB   = (float*) (ws + OFF_VB);
  float*  part = (float*) (ws + OFF_PART);

  // 1) pack weights + biases to bf16 fragment layout
  pack_kernel<<<(NSET * 4 * 8 * 32 * 16 + 255) / 256, 256, 0, stream>>>(
      g_W1, fs_W1, g_W2, fs_W2, g_b1, fs_b1, g_b2, fs_b2, w1p, w2p, b1p, b2p);

  // 2) x = bf16(emb[tok] + pos)
  embed_kernel<<<(ROWS * 64) / 256, 256, 0, stream>>>(tokens, emb, pos_emb, x);

  // 3) V0 = gelu(x@g_W1+b1)@g_W2+b2  -> R   (8192 row-tiles, 8 waves/block)
  mlp_kernel<<<ROWS / 16 / 8, 256, 0, stream>>>(x, w1p, w2p, b1p, b2p, R,
                                                /*s=*/0, /*nOutTiles=*/8, /*outLd=*/128);

  // 4) 12 chord steps: Wm_s = MLP_s(x); V <- mix(V, Wm_s) + V0   (ping-pong)
  for (int s = 1; s <= NW_; ++s) {
    mlp_kernel<<<ROWS / 16 / 8, 256, 0, stream>>>(x, w1p, w2p, b1p, b2p, Wm,
                                                  s, /*nOutTiles=*/1, /*outLd=*/16);
    const float* vin = (s == 1) ? R : ((s & 1) ? VB : VA);
    float*       vout = (s & 1) ? VA : VB;
    chord_kernel<<<ROWS / 8, 256, 0, stream>>>((const float4*)vin, (const float4*)R,
                                               (const float4*)Wm, (float4*)vout);
  }

  // 5) out = V.reshape(B,-1) @ final_W + final_b   (V ends in VB after 12 steps)
  final_partial_kernel<<<dim3(128, B_), 256, 0, stream>>>(VB, final_W, part);
  final_reduce_kernel<<<1, 320, 0, stream>>>(part, final_b, out, 128);
}
